// BaseLoftqLinear_38259568673147
// MI455X (gfx1250) — compile-verified, matrix operand order checked by reading, not run on hardware
//
#include <hip/hip_runtime.h>

// ---------------------------------------------------------------------------
// W4A16 fused quantized-linear + LoRA for gfx1250 (MI455X), wave32 WMMA.
//   y = x @ dequant(qweight, weight_max).T + bias + (x @ lora_A.T) @ lora_B.T
// ---------------------------------------------------------------------------

typedef __attribute__((ext_vector_type(16))) __bf16 v16bf;
typedef __attribute__((ext_vector_type(8)))  float  v8f;

#define IN_F  4096
#define OUT_F 4096
#define RANK  16

#define BM  128     // M tile per workgroup
#define BN  128     // N tile per workgroup
#define KB  64      // K step (== quant block: one scale per row per step)
#define LDX 72      // LDS row stride in bf16 (64 + 8 pad, keeps 16B alignment)
#define LDW 72

// Pack two f32 -> one dword of 2x bf16 (RNE) via the hardware instruction.
// Inline asm is opaque to SLP, so clang cannot re-pair the converts and
// patch the packing with v_perm_b32 (which it does for the cast version).
__device__ __forceinline__ unsigned int packcvt(float a, float b) {
  unsigned int r;
  asm("v_cvt_pk_bf16_f32 %0, %1, %2" : "=v"(r) : "v"(a), "v"(b));
  return r;
}

// Fragment <-> memory views: pure register aliasing, no per-element moves.
union VBF {
  v16bf v;
  float4 f4[2];
  unsigned int u[8];
};

// Wave32 16x32 bf16 A/B fragment from an LDS row: two 16B chunks, bitcast.
__device__ __forceinline__ v16bf ld_frag_lds(const __bf16* row, int c0, int c1) {
  VBF r;
  r.f4[0] = *(const float4*)(row + c0);
  r.f4[1] = *(const float4*)(row + c1);
  return r.v;
}

// Fragment straight from global f32 with pairwise convert (LoRA-A kernel).
__device__ __forceinline__ v16bf ld_frag_g(const float* p, int c0, int c1) {
  float4 f0 = *(const float4*)(p + c0);
  float4 f1 = *(const float4*)(p + c0 + 4);
  float4 f2 = *(const float4*)(p + c1);
  float4 f3 = *(const float4*)(p + c1 + 4);
  VBF r;
  r.u[0] = packcvt(f0.x, f0.y); r.u[1] = packcvt(f0.z, f0.w);
  r.u[2] = packcvt(f1.x, f1.y); r.u[3] = packcvt(f1.z, f1.w);
  r.u[4] = packcvt(f2.x, f2.y); r.u[5] = packcvt(f2.z, f2.w);
  r.u[6] = packcvt(f3.x, f3.y); r.u[7] = packcvt(f3.z, f3.w);
  return r.v;
}

// Dequant 4 packed int32 (2 codes each, low byte) -> 4 dwords of 2x bf16.
// w = table[c]*s = (c*(2/15)-1)*s = fma(c, 2s/15, -s)
__device__ __forceinline__ void dq8(int4 q, float s2, float ns,
                                    unsigned int* o) {
  int a[4] = { q.x, q.y, q.z, q.w };
#pragma unroll
  for (int i = 0; i < 4; ++i) {
    float lo = __builtin_fmaf((float)(a[i] & 15), s2, ns);
    float hi = __builtin_fmaf((float)((a[i] >> 4) & 15), s2, ns);
    o[i] = packcvt(lo, hi);
  }
}

// ---------------------------------------------------------------------------
// Kernel 1: a[M, RANK] = x @ lora_A.T   (one 16x16 output tile per wave)
// ---------------------------------------------------------------------------
__global__ __launch_bounds__(256) void lora_a_kernel(
    const float* __restrict__ x, const float* __restrict__ lora_A,
    float* __restrict__ a_out, int M) {
  const int lane = threadIdx.x & 31;
  const int wv   = (blockIdx.x * 256 + (int)threadIdx.x) >> 5;
  const int m0   = wv * 16;
  if (m0 >= M) return;  // wave-uniform: EXEC all-1s at every WMMA
  const int half = lane >> 4;
  const int l15  = lane & 15;

  const float* xr = x + (size_t)(m0 + l15) * IN_F;  // A row: M = lane&15
  const float* ar = lora_A + (size_t)l15 * IN_F;    // B col: N(=r) = lane&15

  v8f acc = {};
  for (int k0 = 0; k0 < IN_F; k0 += 32) {
    const int c0 = k0 + half * 8;
    const int c1 = c0 + 16;
    v16bf af = ld_frag_g(xr, c0, c1);
    v16bf bf = ld_frag_g(ar, c0, c1);
    acc = __builtin_amdgcn_wmma_f32_16x16x32_bf16(false, af, false, bf,
                                                  (short)0, acc, false, false);
  }
  // D layout: VGPR j -> M = j + 8*half, N = lane&15
#pragma unroll
  for (int j = 0; j < 8; ++j)
    a_out[(size_t)(m0 + half * 8 + j) * RANK + l15] = acc[j];
}

// ---------------------------------------------------------------------------
// Kernel 2: fused dequant-GEMM + bias + LoRA epilogue.
// 256 threads = 8 waves arranged 2(M) x 4(N); each wave: 64M x 32N
// = 8 accumulators; per K-step: 4 A-frags + 2 B-frags -> 16 WMMA.
// Global loads are register double-buffered: tile k+1 loads issue before the
// tile-k compute phase, so s_wait_loadcnt lands after the 16 WMMAs.
// ---------------------------------------------------------------------------
__global__ __launch_bounds__(256) void wqlinear_kernel(
    const float* __restrict__ x, const int* __restrict__ qweight,
    const float* __restrict__ wmax, const float* __restrict__ lora_B,
    const float* __restrict__ bias, const float* __restrict__ a_ws,
    float* __restrict__ out) {
  __shared__ __bf16 xs[BM * LDX];   // 18.4 KB: x tile bf16, [m][k]
  __shared__ __bf16 wsh[BN * LDW];  // 18.4 KB: dequant W tile bf16, [n][k]

  const int t     = threadIdx.x;
  const int lane  = t & 31;
  const int wave  = t >> 5;
  const int waveM = wave >> 2;  // 0..1 -> 64-row band
  const int waveN = wave & 3;   // 0..3 -> 32-col band
  const int half  = lane >> 4;
  const int l15   = lane & 15;

  const int blockN = blockIdx.x * BN;
  const int blockM = blockIdx.y * BM;

  // cooperative tile loads: 2 threads per row (x: 32 f32, W: 16 int32 each)
  const int    xrow = t >> 1;
  const int    xcol = (t & 1) * 32;
  const float* xg   = x + (size_t)(blockM + xrow) * IN_F + xcol;
  const int    gn   = blockN + xrow;
  const int*   qg   = qweight + (((size_t)gn * IN_F + xcol) >> 1);
  const float* smax = wmax + (size_t)gn * (IN_F / 64);

  v8f acc[8] = {};  // [tt(0..3)][nn(0..1)] -> idx = tt*2+nn

  // ---- pipeline registers + loader ----
  float4 xf[8];
  int4   qf[4];
  float  s;
  auto load_global = [&](int k0) {
    const float4* p = (const float4*)(xg + k0);
#pragma unroll
    for (int i = 0; i < 8; ++i) xf[i] = p[i];
    const int4* qp = (const int4*)(qg + (k0 >> 1));
#pragma unroll
    for (int i = 0; i < 4; ++i) qf[i] = qp[i];
    s = smax[k0 >> 6];  // one 64-wide quant block per step
  };

  load_global(0);

  for (int k0 = 0; k0 < IN_F; k0 += KB) {
    // convert current tile in registers (overlaps other waves' compute)
    unsigned int xo[16];
#pragma unroll
    for (int i = 0; i < 8; ++i) {
      xo[2 * i]     = packcvt(xf[i].x, xf[i].y);
      xo[2 * i + 1] = packcvt(xf[i].z, xf[i].w);
    }
    const float  s2 = s * (2.0f / 15.0f);
    const float  ns = -s;
    unsigned int wo[16];
    dq8(qf[0], s2, ns, wo + 0);  dq8(qf[1], s2, ns, wo + 4);
    dq8(qf[2], s2, ns, wo + 8);  dq8(qf[3], s2, ns, wo + 12);

    __syncthreads();  // previous compute done before LDS overwrite
    {
      uint4* dx = (uint4*)&xs[xrow * LDX + xcol];
      uint4* dw = (uint4*)&wsh[xrow * LDW + xcol];
#pragma unroll
      for (int i = 0; i < 4; ++i) {
        dx[i] = make_uint4(xo[4 * i], xo[4 * i + 1], xo[4 * i + 2], xo[4 * i + 3]);
        dw[i] = make_uint4(wo[4 * i], wo[4 * i + 1], wo[4 * i + 2], wo[4 * i + 3]);
      }
    }
    __syncthreads();

    if (k0 + KB < IN_F) load_global(k0 + KB);  // in flight during WMMAs

    // ---- 2 K-subtiles x (2 B-frags + 4 A-frags -> 8 WMMA) ----
#pragma unroll
    for (int ks = 0; ks < 2; ++ks) {
      const int c0 = ks * 32 + half * 8;
      const int c1 = c0 + 16;
      v16bf bfr[2];
#pragma unroll
      for (int nn = 0; nn < 2; ++nn)
        bfr[nn] = ld_frag_lds(&wsh[(waveN * 32 + nn * 16 + l15) * LDW], c0, c1);
#pragma unroll
      for (int tt = 0; tt < 4; ++tt) {
        v16bf af = ld_frag_lds(&xs[(waveM * 64 + tt * 16 + l15) * LDX], c0, c1);
#pragma unroll
        for (int nn = 0; nn < 2; ++nn)
          acc[tt * 2 + nn] = __builtin_amdgcn_wmma_f32_16x16x32_bf16(
              false, af, false, bfr[nn], (short)0, acc[tt * 2 + nn], false,
              false);
      }
    }
  }

  // ---- epilogue: + bias + rank-16 LoRA, f32 ----
  float  bv[2];
  float4 lb[2][4];
#pragma unroll
  for (int nn = 0; nn < 2; ++nn) {
    const int n = blockN + waveN * 32 + nn * 16 + l15;
    bv[nn] = bias[n];
    const float4* lbp = (const float4*)(lora_B + (size_t)n * RANK);
#pragma unroll
    for (int i = 0; i < 4; ++i) lb[nn][i] = lbp[i];
  }
#pragma unroll
  for (int tt = 0; tt < 4; ++tt) {
#pragma unroll
    for (int j = 0; j < 8; ++j) {
      const int m = blockM + waveM * 64 + tt * 16 + half * 8 + j;
      const float4* ap = (const float4*)(a_ws + (size_t)m * RANK);
      const float4 a0 = ap[0], a1 = ap[1], a2 = ap[2], a3 = ap[3];
#pragma unroll
      for (int nn = 0; nn < 2; ++nn) {
        const int n = blockN + waveN * 32 + nn * 16 + l15;
        float v = acc[tt * 2 + nn][j] + bv[nn];
        v += a0.x * lb[nn][0].x + a0.y * lb[nn][0].y +
             a0.z * lb[nn][0].z + a0.w * lb[nn][0].w;
        v += a1.x * lb[nn][1].x + a1.y * lb[nn][1].y +
             a1.z * lb[nn][1].z + a1.w * lb[nn][1].w;
        v += a2.x * lb[nn][2].x + a2.y * lb[nn][2].y +
             a2.z * lb[nn][2].z + a2.w * lb[nn][2].w;
        v += a3.x * lb[nn][3].x + a3.y * lb[nn][3].y +
             a3.z * lb[nn][3].z + a3.w * lb[nn][3].w;
        out[(size_t)m * OUT_F + n] = v;
      }
    }
  }
}

// ---------------------------------------------------------------------------
extern "C" void kernel_launch(void* const* d_in, const int* in_sizes, int n_in,
                              void* d_out, int out_size, void* d_ws,
                              size_t ws_size, hipStream_t stream) {
  const float* x       = (const float*)d_in[0];
  const int*   qweight = (const int*)d_in[1];
  const float* wmax    = (const float*)d_in[2];
  const float* lora_A  = (const float*)d_in[3];
  const float* lora_B  = (const float*)d_in[4];
  const float* bias    = (const float*)d_in[5];
  float*       out     = (float*)d_out;
  float*       a_ws    = (float*)d_ws;  // [M, RANK] f32 scratch (512 KB)

  const int M = in_sizes[0] / IN_F;  // 8192 = 4 * 2048

  // 1) a = x @ lora_A.T : one wave per 16 rows
  const int waves  = (M + 15) / 16;
  const int blocks = (waves + 7) / 8;
  lora_a_kernel<<<blocks, 256, 0, stream>>>(x, lora_A, a_ws, M);

  // 2) fused dequant GEMM + bias + LoRA
  dim3 grid(OUT_F / BN, M / BM);  // 32 x 64 = 2048 workgroups
  wqlinear_kernel<<<grid, 256, 0, stream>>>(x, qweight, wmax, lora_B, bias,
                                            a_ws, out);
}
